// RNN_44590350467352
// MI455X (gfx1250) — matrix operand order repaired and checked by reading, not compiled, not run on hardware
//
#include <hip/hip_runtime.h>

// RNN: h_t = tanh(x_t U + h_{t-1} W + b), B=256 T=512 D=K=256, out = h_T [256,1,256] f32
// (1) transpose W,U -> bf16 [N][K]; (2) parallel WMMA GEMM xU+b -> ws f32 [T][B][K];
// (3) sequential scan: W fragments register-resident, h in 8KB conflict-free LDS layout.

#define Tsz 512

typedef __bf16 bf16;
typedef __attribute__((ext_vector_type(16))) __bf16 v16bf;
typedef __attribute__((ext_vector_type(8)))  float  v8f;

union BF16x16 { uint4 u[2]; v16bf v; };

__device__ __forceinline__ bf16 f2bf(float f) {
  unsigned u = __builtin_bit_cast(unsigned, f);
  u += 0x7FFFu + ((u >> 16) & 1u);            // round-to-nearest-even
  unsigned short hs = (unsigned short)(u >> 16);
  return __builtin_bit_cast(bf16, hs);
}

__device__ __forceinline__ float fast_tanh(float x) {
#if __has_builtin(__builtin_amdgcn_tanhf)
  return __builtin_amdgcn_tanhf(x);           // v_tanh_f32 (gfx1250 trans op)
#else
  x = fminf(10.0f, fmaxf(-10.0f, x));
  float e = __builtin_amdgcn_exp2f(x * 2.8853900817779268f);   // exp(2x)
  return (e - 1.0f) * __builtin_amdgcn_rcpf(e + 1.0f);
#endif
}

// ---------------- Kernel 0: W,U -> bf16 transposed [N][K] ----------------
__global__ void prep_transpose(const float* __restrict__ U, const float* __restrict__ W,
                               bf16* __restrict__ UT, bf16* __restrict__ WT) {
  int n = blockIdx.x;
  int k = threadIdx.x;
  UT[n * 256 + k] = f2bf(U[k * 256 + n]);
  WT[n * 256 + k] = f2bf(W[k * 256 + n]);
}

// ---------------- Kernel 1: xU + b (f32 out, [T][B][K] layout) ----------------
__global__ void xu_gemm(const float* __restrict__ X, const bf16* __restrict__ UT,
                        const float* __restrict__ bias, float* __restrict__ xu) {
  __shared__ __align__(16) bf16 Ash[16 * 256];   // A tile, bf16
  const int tid  = threadIdx.x;
  const int wave = tid >> 5, lane = tid & 31, g = lane >> 4, ln = lane & 15;
  const int mt   = blockIdx.x;                   // 8192 tiles of 16 rows over [B*T, D]

  { // stage + convert A tile (16x256 f32 -> bf16 in LDS)
    int row = tid >> 4;
    int cb  = (tid & 15) << 4;
    const float* src = X + (size_t)(mt * 16 + row) * 256 + cb;
    BF16x16 tv;
    #pragma unroll
    for (int i = 0; i < 16; ++i) tv.v[i] = f2bf(src[i]);
    *(uint4*)(Ash + row * 256 + cb)     = tv.u[0];
    *(uint4*)(Ash + row * 256 + cb + 8) = tv.u[1];
  }
  __syncthreads();

  const int nt0 = wave * 2, nt1 = nt0 + 1;
  const int n0 = nt0 * 16 + ln, n1 = nt1 * 16 + ln;

  v8f acc0 = {}, acc1 = {};
  #pragma unroll
  for (int kk = 0; kk < 8; ++kk) {
    BF16x16 a, b0, b1;
    const bf16* arow = Ash + ln * 256 + kk * 32;          // A: lane=M, contiguous K=16g..16g+15
    a.u[0] = *(const uint4*)(arow + g * 16);
    a.u[1] = *(const uint4*)(arow + g * 16 + 8);
    const bf16* b0p = UT + n0 * 256 + kk * 32 + g * 16;   // B: lane=N, contiguous K=16g..16g+15
    b0.u[0] = *(const uint4*)(b0p);
    b0.u[1] = *(const uint4*)(b0p + 8);
    const bf16* b1p = UT + n1 * 256 + kk * 32 + g * 16;
    b1.u[0] = *(const uint4*)(b1p);
    b1.u[1] = *(const uint4*)(b1p + 8);
    acc0 = __builtin_amdgcn_wmma_f32_16x16x32_bf16(false, a.v, false, b0.v, (short)0, acc0, false, false);
    acc1 = __builtin_amdgcn_wmma_f32_16x16x32_bf16(false, a.v, false, b1.v, (short)0, acc1, false, false);
  }

  float bias0 = bias[n0], bias1 = bias[n1];
  #pragma unroll
  for (int r = 0; r < 8; ++r) {
    int m   = r + 8 * g;                   // C layout: VGPR r -> M = r + 8*(lane>=16)
    int row = mt * 16 + m;                 // flattened b*T + t
    int b_  = row >> 9;                    // /512
    int t_  = row & 511;
    size_t o = ((size_t)t_ * 256 + b_) * 256;
    xu[o + n0] = acc0[r] + bias0;
    xu[o + n1] = acc1[r] + bias1;
  }
}

// ---------------- Kernel 2: sequential scan ----------------
// h stored in LDS in A-fragment-ready layout: [kk:8][chunk:2][lane:32][16B] = 8KB.
// Reader (lane L, kstep kk): chunk0 = hA4[kk*64 + L], chunk1 = hA4[kk*64 + 32 + L]
//   -> lanes 0-15 sweep all 64 banks: conflict-free ds_load_b128.
// Writer: element h(m, k) lives at byte (k>>5)*1024 + ((k>>3)&1)*512 + ((k>>4)&1)*256 + m*16 + (k&7)*2.
__global__ void rnn_scan(const float* __restrict__ xu, const bf16* __restrict__ WT,
                         float* __restrict__ out) {
  __shared__ __align__(16) char hA[8192];

  const int tid  = threadIdx.x;
  const int wave = tid >> 5, lane = tid & 31, g = lane >> 4, ln = lane & 15;
  const int b0   = blockIdx.x * 16;             // 16 blocks x 16 batch rows

  const int nt0 = wave * 2, nt1 = nt0 + 1;
  const int n0 = nt0 * 16 + ln, n1 = nt1 * 16 + ln;

  // Hoist all W fragments for this wave into VGPRs (invariant across the 512 steps).
  v16bf wf0[8], wf1[8];
  #pragma unroll
  for (int kk = 0; kk < 8; ++kk) {
    BF16x16 t0, t1;
    const bf16* p0 = WT + n0 * 256 + kk * 32 + g * 16;
    t0.u[0] = *(const uint4*)(p0);
    t0.u[1] = *(const uint4*)(p0 + 8);
    wf0[kk] = t0.v;
    const bf16* p1 = WT + n1 * 256 + kk * 32 + g * 16;
    t1.u[0] = *(const uint4*)(p1);
    t1.u[1] = *(const uint4*)(p1 + 8);
    wf1[kk] = t1.v;
  }

  { // h0 = 0
    uint4 z = {};
    uint4* hz = (uint4*)hA;
    for (int i = tid; i < 8192 / 16; i += 256) hz[i] = z;
  }
  __syncthreads();

  // per-lane scatter bases for writing h (k = n0 / n1 fixed per lane+tile)
  const int sb0 = ((n0 >> 5) << 10) + (((n0 >> 3) & 1) << 9) + (((n0 >> 4) & 1) << 8) + ((n0 & 7) << 1);
  const int sb1 = ((n1 >> 5) << 10) + (((n1 >> 3) & 1) << 9) + (((n1 >> 4) & 1) << 8) + ((n1 & 7) << 1);

  float px0[8], px1[8];                          // xU prefetch registers
  {
    const float* base = xu + (size_t)b0 * 256;   // t = 0
    #pragma unroll
    for (int r = 0; r < 8; ++r) {
      int m = r + 8 * g;
      px0[r] = base[m * 256 + n0];
      px1[r] = base[m * 256 + n1];
    }
  }

  const uint4* hA4 = (const uint4*)hA;

  for (int t = 0; t < Tsz; ++t) {
    v8f acc0, acc1;
    #pragma unroll
    for (int r = 0; r < 8; ++r) { acc0[r] = px0[r]; acc1[r] = px1[r]; }

    if (t + 1 < Tsz) {                           // prefetch next step's xU during WMMAs
      const float* base = xu + ((size_t)(t + 1) * 256 + b0) * 256;
      #pragma unroll
      for (int r = 0; r < 8; ++r) {
        int m = r + 8 * g;
        px0[r] = base[m * 256 + n0];
        px1[r] = base[m * 256 + n1];
      }
    }

    #pragma unroll
    for (int kk = 0; kk < 8; ++kk) {
      BF16x16 a;
      a.u[0] = hA4[kk * 64 + lane];
      a.u[1] = hA4[kk * 64 + 32 + lane];
      acc0 = __builtin_amdgcn_wmma_f32_16x16x32_bf16(false, a.v, false, wf0[kk], (short)0, acc0, false, false);
      acc1 = __builtin_amdgcn_wmma_f32_16x16x32_bf16(false, a.v, false, wf1[kk], (short)0, acc1, false, false);
    }

    float h0[8], h1[8];
    #pragma unroll
    for (int r = 0; r < 8; ++r) { h0[r] = fast_tanh(acc0[r]); h1[r] = fast_tanh(acc1[r]); }

    __syncthreads();                             // all A-frag reads done before overwrite

    #pragma unroll
    for (int r = 0; r < 8; ++r) {
      int m = r + 8 * g;
      *(bf16*)(hA + sb0 + m * 16) = f2bf(h0[r]);
      *(bf16*)(hA + sb1 + m * 16) = f2bf(h1[r]);
    }
    if (t == Tsz - 1) {
      #pragma unroll
      for (int r = 0; r < 8; ++r) {
        int m = r + 8 * g;
        out[(size_t)(b0 + m) * 256 + n0] = h0[r];
        out[(size_t)(b0 + m) * 256 + n1] = h1[r];
      }
    }
    __syncthreads();                             // h visible before next step's reads
  }
}

extern "C" void kernel_launch(void* const* d_in, const int* in_sizes, int n_in,
                              void* d_out, int out_size, void* d_ws, size_t ws_size,
                              hipStream_t stream) {
  const float* X    = (const float*)d_in[0];   // [256,512,256]
  const float* U    = (const float*)d_in[1];   // [256,256]
  const float* W    = (const float*)d_in[2];   // [256,256]
  const float* bias = (const float*)d_in[3];   // [256]
  float* out = (float*)d_out;

  char* ws = (char*)d_ws;
  float* xu = (float*)ws;                              // 512*256*256*4 = 128MB, [T][B][K]
  bf16* WT  = (bf16*)(ws + (size_t)512 * 256 * 256 * 4);
  bf16* UT  = WT + 256 * 256;

  prep_transpose<<<256, 256, 0, stream>>>(U, W, UT, WT);
  xu_gemm<<<8192, 256, 0, stream>>>(X, UT, bias, xu);
  rnn_scan<<<16, 256, 0, stream>>>(xu, WT, out);
}
